// Pointnet2DetHead_27582279975295
// MI455X (gfx1250) — compile-verified, matrix operand order checked by reading, not run on hardware
//
#include <hip/hip_runtime.h>
#include <math.h>

// Problem constants (from reference)
#define N_PTS   100000
#define P_PROP  256
#define R_FEAT  256
#define C_CLS   21            // C + 1
#define NCHUNK  3125          // N_PTS / 32 (exact)
#define PTILES  16            // P_PROP / 16 : one wave accumulates all proposal tiles
#define NWSPLIT 192           // N splits per 16-col R-group -> 16*192 = 3072 worker waves

// Exact compensation for f16-denormal A trick: A bit value = 2^-24
#define A_SCALE 16777216.0f   // 2^24

typedef __attribute__((ext_vector_type(2)))  float    v2f;
typedef __attribute__((ext_vector_type(8)))  float    v8f;
typedef __attribute__((ext_vector_type(8)))  unsigned v8u;
typedef __attribute__((ext_vector_type(16))) _Float16 v16h;

// Workspace layout (bytes)
//   [0,       1024)    countsF : 256 f32 (padded)
//   [1024,    263168)  sums    : 256x256 f32 (scaled by 2^-24)
//   [263168,  525312)  roi     : 256x256 f32
//   [525312,  ...)     mask    : NCHUNK x 256 u32 (bit i = point chunk*32+i inside box p)
#define WS_SUMS_OFF  1024
#define WS_ROI_OFF   (1024 + 262144)
#define WS_MASK_OFF  (1024 + 2*262144)

// ---------------------------------------------------------------------------
// Kernel 0: zero counts + sums (65792 contiguous floats)
// ---------------------------------------------------------------------------
__global__ __launch_bounds__(256) void zero_kernel(float* __restrict__ p, int n) {
  int i = blockIdx.x * 256 + threadIdx.x;
  if (i < n) p[i] = 0.0f;
}

// ---------------------------------------------------------------------------
// Kernel 1: point-in-box mask bits + per-proposal counts.
// One wave = 32 consecutive points (one mask word per proposal).
// mask layout: mask[chunk*256 + p]  (coalesced for the GEMM kernel)
// ---------------------------------------------------------------------------
__global__ __launch_bounds__(256) void mask_kernel(const float* __restrict__ proposals,
                                                   const float* __restrict__ xyz,
                                                   unsigned* __restrict__ mask,
                                                   float* __restrict__ countsF) {
  __shared__ float    slo[P_PROP][3];
  __shared__ float    shi[P_PROP][3];
  __shared__ unsigned scnt[P_PROP];

  const int tid = threadIdx.x;
  if (tid < P_PROP) {
    float cx = proposals[tid * 6 + 0], cy = proposals[tid * 6 + 1], cz = proposals[tid * 6 + 2];
    float hx = proposals[tid * 6 + 3] * 0.5f;
    float hy = proposals[tid * 6 + 4] * 0.5f;
    float hz = proposals[tid * 6 + 5] * 0.5f;
    slo[tid][0] = cx - hx; slo[tid][1] = cy - hy; slo[tid][2] = cz - hz;
    shi[tid][0] = cx + hx; shi[tid][1] = cy + hy; shi[tid][2] = cz + hz;
    scnt[tid]   = 0u;
  }
  __syncthreads();

  const int n     = blockIdx.x * 256 + tid;
  const int lane  = tid & 31;
  const int chunk = blockIdx.x * 8 + (tid >> 5);   // global wave id == 32-point chunk

  float x = -1e30f, y = -1e30f, z = -1e30f;
  if (n < N_PTS) { x = xyz[n * 3 + 0]; y = xyz[n * 3 + 1]; z = xyz[n * 3 + 2]; }

  if (chunk < NCHUNK) {
    for (int pb = 0; pb < 8; pb++) {              // 8 groups of 32 proposals
      unsigned myw = 0u;
      for (int i = 0; i < 32; i++) {
        int p = pb * 32 + i;
        bool inside = (x >= slo[p][0]) & (x <= shi[p][0]) &
                      (y >= slo[p][1]) & (y <= shi[p][1]) &
                      (z >= slo[p][2]) & (z <= shi[p][2]);
        unsigned w = (unsigned)__ballot(inside);  // wave32: low 32 bits valid
        if (lane == i) myw = w;                   // lane i keeps word of proposal pb*32+i
      }
      mask[chunk * P_PROP + pb * 32 + lane] = myw;       // coalesced 128B store
      atomicAdd(&scnt[pb * 32 + lane], (unsigned)__popc(myw));
    }
  }
  __syncthreads();
  if (tid < P_PROP)
    __hip_atomic_fetch_add(&countsF[tid], (float)scnt[tid],
                           __ATOMIC_RELAXED, __HIP_MEMORY_SCOPE_AGENT);
}

// ---------------------------------------------------------------------------
// A-operand expansion: mask bits -> f16 *denormals* (bit pattern 0x0001 ==
// 2^-24 exactly). 3 VALU per packed dword (bfe, bfe, lshl_or). The exact
// power-of-two scale is compensated in the head kernel (roi * 2^24).
// Dword d of the 16-bit A layout holds K pair 16*(d>>2) + 8*half + 2*(d&3).
// ---------------------------------------------------------------------------
__device__ __forceinline__ v8u expand_bits(unsigned w, int half) {
  v8u au;
#pragma unroll
  for (int d = 0; d < 8; d++) {
    const int k0 = 16 * (d >> 2) + 8 * half + 2 * (d & 3);
    unsigned b0 = (w >> k0) & 1u;
    unsigned b1 = (w >> (k0 + 1)) & 1u;
    au[d] = b0 | (b1 << 16);
  }
  return au;
}

// ---------------------------------------------------------------------------
// Kernel 2: sums(P,R) += 2^-24 * maskf(P,N) @ f16(seg_feats)(N,R) via
// V_WMMA_F32_16X16X32_F16. One wave: 16 R-columns x all 16 proposal tiles
// (128 acc VGPRs, stays under the 256-VGPR window -> no s_set_vgpr_msb).
// K=32 per WMMA == one whole 32-point chunk per instruction. Two alternating
// A buffers keep >=25 slots between a WMMA and the rewrite of its A operand
// (avoids the 4-coexec WMMA->VALU WAR hazard nops).
// ---------------------------------------------------------------------------
__global__ __launch_bounds__(256) void gemm_kernel(const float* __restrict__ seg,
                                                   const unsigned* __restrict__ mask,
                                                   float* __restrict__ sums) {
  const int lane  = threadIdx.x & 31;
  const int wave  = blockIdx.x * 8 + (threadIdx.x >> 5);
  const int rt    = wave & 15;              // 16-column R group
  const int split = wave >> 4;              // N split 0..NWSPLIT-1
  const int row16 = lane & 15;
  const int half  = lane >> 4;              // K-half selector for 16-bit layouts
  const int col   = rt * 16 + row16;

  v8f acc[PTILES];
#pragma unroll
  for (int t = 0; t < PTILES; t++)
    acc[t] = (v8f){0.f, 0.f, 0.f, 0.f, 0.f, 0.f, 0.f, 0.f};

  for (int c = split; c < NCHUNK; c += NWSPLIT) {
    const int n0 = c * 32;

    // mask words: lanes 0-15 read a contiguous 64B run per tile
    unsigned w[PTILES];
#pragma unroll
    for (int t = 0; t < PTILES; t++)
      w[t] = mask[c * P_PROP + t * 16 + row16];

    // B tile (32x16 f16, K x N): dword d holds K pair 16*(d>>2)+8*half+2*(d&3)
    v16h B;
#pragma unroll
    for (int d = 0; d < 8; d++) {
      const int k0 = 16 * (d >> 2) + 8 * half + 2 * (d & 3);
      B[2 * d]     = (_Float16)seg[(n0 + k0) * R_FEAT + col];
      B[2 * d + 1] = (_Float16)seg[(n0 + k0 + 1) * R_FEAT + col];
    }

    // A tiles: software-pipelined double buffer
    v8u abuf[2];
    abuf[0] = expand_bits(w[0], half);
    abuf[1] = expand_bits(w[1], half);
#pragma unroll
    for (int t = 0; t < PTILES; t++) {
      acc[t] = __builtin_amdgcn_wmma_f32_16x16x32_f16(
          false, __builtin_bit_cast(v16h, abuf[t & 1]), false, B,
          (short)0, acc[t], false, false);
      if (t + 2 < PTILES) abuf[t & 1] = expand_bits(w[t + 2], half);
    }
  }

  // C tile: VGPR j holds rows j (lanes 0-15) and j+8 (lanes 16-31)
#pragma unroll
  for (int t = 0; t < PTILES; t++)
#pragma unroll
    for (int j = 0; j < 8; j++) {
      int p = t * 16 + j + 8 * half;
      __hip_atomic_fetch_add(&sums[p * R_FEAT + col], acc[t][j],
                             __ATOMIC_RELAXED, __HIP_MEMORY_SCOPE_AGENT);
    }
}

// ---------------------------------------------------------------------------
// Kernel 3 (single workgroup, 32 waves): roi = 2^24 * sums / max(counts,1);
// logits = roi @ W + b (exact f32 WMMA); cls row-softmax; obj column-softmax.
// ---------------------------------------------------------------------------
__global__ __launch_bounds__(1024) void head_kernel(const float* __restrict__ sums,
                                                    const float* __restrict__ countsF,
                                                    float* __restrict__ roi,
                                                    const float* __restrict__ Wc,
                                                    const float* __restrict__ bc,
                                                    const float* __restrict__ Wo,
                                                    const float* __restrict__ bo,
                                                    float* __restrict__ out) {
  __shared__ float lg[2 * P_PROP * 22];   // [head][p][c], c padded to 22
  const int tid = threadIdx.x;

  // 1) roi pooling normalize (undo the exact 2^-24 A-operand scale here)
  for (int idx = tid; idx < P_PROP * R_FEAT; idx += 1024) {
    int p = idx >> 8;
    roi[idx] = sums[idx] * A_SCALE / fmaxf(countsF[p], 1.0f);
  }
  __threadfence_block();
  __syncthreads();

  // 2) two 256x256 @ 256x21 GEMMs: 64 jobs = head(2) x ptile(16) x ctile(2)
  const int wave = tid >> 5, lane = tid & 31;
  const int row16 = lane & 15, half = lane >> 4, koff = 2 * half;
  for (int jj = 0; jj < 2; jj++) {
    const int job  = wave + jj * 32;
    const int head = job >> 5;
    const int rem  = job & 31;
    const int pt   = rem >> 1, ct = rem & 1;
    const float* W    = head ? Wo : Wc;
    const float* bias = head ? bo : bc;
    const int  c  = ct * 16 + row16;
    const bool cv = (c < C_CLS);
    const int  ar = pt * 16 + row16;
    v8f acc = (v8f){0.f, 0.f, 0.f, 0.f, 0.f, 0.f, 0.f, 0.f};
#pragma unroll 4
    for (int k = 0; k < R_FEAT; k += 4) {
      v2f A, B;
      A.x = roi[ar * R_FEAT + k + koff];
      A.y = roi[ar * R_FEAT + k + koff + 1];
      B.x = cv ? W[(k + koff) * C_CLS + c] : 0.0f;
      B.y = cv ? W[(k + koff + 1) * C_CLS + c] : 0.0f;
      acc = __builtin_amdgcn_wmma_f32_16x16x4_f32(
          false, A, false, B, (short)0, acc, false, false);
    }
    if (cv) {
      float bv = bias[c];
#pragma unroll
      for (int j = 0; j < 8; j++) {
        int p = pt * 16 + j + 8 * half;
        lg[head * P_PROP * 22 + p * 22 + c] = acc[j] + bv;
      }
    }
  }
  __syncthreads();

  // 3a) cls softmax over axis=1 (one thread per proposal row)
  if (tid < P_PROP) {
    float* row = &lg[tid * 22];
    float m = row[0];
    for (int c = 1; c < C_CLS; c++) m = fmaxf(m, row[c]);
    float s = 0.f;
    for (int c = 0; c < C_CLS; c++) { float e = expf(row[c] - m); row[c] = e; s += e; }
    float inv = 1.0f / s;
    for (int c = 0; c < C_CLS; c++) row[c] *= inv;
  }
  // 3b) obj softmax over axis=0 (one wave per class column, shfl reductions)
  if (wave < C_CLS) {
    const int c = wave;
    float v[8];
    float m = -1e30f;
#pragma unroll
    for (int j = 0; j < 8; j++) {
      int p = lane + 32 * j;
      v[j] = lg[P_PROP * 22 + p * 22 + c];
      m = fmaxf(m, v[j]);
    }
    for (int off = 16; off >= 1; off >>= 1) m = fmaxf(m, __shfl_xor(m, off, 32));
    float s = 0.f;
#pragma unroll
    for (int j = 0; j < 8; j++) { v[j] = expf(v[j] - m); s += v[j]; }
    for (int off = 16; off >= 1; off >>= 1) s += __shfl_xor(s, off, 32);
    float inv = 1.0f / s;
#pragma unroll
    for (int j = 0; j < 8; j++) {
      int p = lane + 32 * j;
      lg[P_PROP * 22 + p * 22 + c] = v[j] * inv;
    }
  }
  __syncthreads();

  // 4) out = cls_p * obj_p  (256x21 row-major)
  for (int idx = tid; idx < P_PROP * C_CLS; idx += 1024) {
    int p = idx / C_CLS, c = idx % C_CLS;
    out[idx] = lg[p * 22 + c] * lg[P_PROP * 22 + p * 22 + c];
  }
}

// ---------------------------------------------------------------------------
extern "C" void kernel_launch(void* const* d_in, const int* in_sizes, int n_in,
                              void* d_out, int out_size, void* d_ws, size_t ws_size,
                              hipStream_t stream) {
  const float* proposals = (const float*)d_in[0];
  const float* xyz       = (const float*)d_in[1];
  const float* seg       = (const float*)d_in[2];
  const float* Wc        = (const float*)d_in[3];
  const float* bc        = (const float*)d_in[4];
  const float* Wo        = (const float*)d_in[5];
  const float* bo        = (const float*)d_in[6];
  float*       out       = (float*)d_out;

  char*     ws      = (char*)d_ws;
  float*    countsF = (float*)ws;
  float*    sums    = (float*)(ws + WS_SUMS_OFF);
  float*    roi     = (float*)(ws + WS_ROI_OFF);
  unsigned* mask    = (unsigned*)(ws + WS_MASK_OFF);

  // counts + sums are contiguous: 65792 floats
  zero_kernel<<<257, 256, 0, stream>>>(countsF, 65792);
  mask_kernel<<<(N_PTS + 255) / 256, 256, 0, stream>>>(proposals, xyz, mask, countsF);
  gemm_kernel<<<(16 * NWSPLIT) / 8, 256, 0, stream>>>(seg, mask, sums);
  head_kernel<<<1, 1024, 0, stream>>>(sums, countsF, roi, Wc, bc, Wo, bo, out);
}
